// TransformerBlock_82265803587775
// MI455X (gfx1250) — matrix operand order repaired and checked by reading, not compiled, not run on hardware
//
#include <hip/hip_runtime.h>
#include <cstddef>
#include <cstdint>

// ---------------------------------------------------------------- constants
#define T_DIM   512
#define BATCH   32
#define D_DIM   1024
#define H_DIM   16
#define DFF_DIM 4096
#define DK_DIM  64
#define NROWS   (T_DIM * BATCH)     // 16384 token rows
#define SCALE_F 0.125f              // 1/sqrt(64)
#define EPS_F   1e-5f

// ---------------------------------------------------------------- CDNA5 feature detection
#if defined(__AMDGCN__) && __has_builtin(__builtin_amdgcn_global_load_async_to_lds_b128)
#define USE_ASYNC 1
#else
#define USE_ASYNC 0
#endif
#if defined(__AMDGCN__) && __has_builtin(__builtin_amdgcn_tensor_load_to_lds)
#define USE_TDM 1
#else
#define USE_TDM 0
#endif

// ---------------------------------------------------------------- types
typedef __attribute__((ext_vector_type(8)))  __bf16 v8bf;
typedef __attribute__((ext_vector_type(16))) __bf16 v16bf;
typedef __attribute__((ext_vector_type(8)))  float  v8f;
typedef __attribute__((ext_vector_type(4)))  int    v4i;
typedef __attribute__((ext_vector_type(8)))  int    v8i;
typedef __attribute__((ext_vector_type(4)))  unsigned int v4u;
// address-space-qualified v4i pointers (typedef so the attribute binds to the
// pointee; a cast-expression attribute gets silently dropped by clang)
typedef __attribute__((address_space(1))) v4i* gv4i_p;
typedef __attribute__((address_space(3))) v4i* lv4i_p;

__device__ __forceinline__ unsigned short f2bf(float f) {
  unsigned int u = __float_as_uint(f);
  u += 0x7fffu + ((u >> 16) & 1u);          // round-to-nearest-even
  return (unsigned short)(u >> 16);
}

// Build a 16-element bf16 fragment from two contiguous 16B chunks.
__device__ __forceinline__ v16bf load_frag16(const unsigned short* p0,
                                             const unsigned short* p1) {
  v8bf a = *(const v8bf*)p0;
  v8bf b = *(const v8bf*)p1;
  v16bf r;
#pragma unroll
  for (int i = 0; i < 8; ++i) { r[i] = a[i]; r[i + 8] = b[i]; }
  return r;
}

__device__ __forceinline__ v8f wmma_bf16(v16bf a, v16bf b, v8f c) {
  return __builtin_amdgcn_wmma_f32_16x16x32_bf16(false, a, false, b,
                                                 (short)0, c, false, false);
}

__device__ __forceinline__ void async_b128(const unsigned short* g,
                                           unsigned short* l) {
#if USE_ASYNC
  __builtin_amdgcn_global_load_async_to_lds_b128((gv4i_p)(g), (lv4i_p)(l), 0, 0);
#else
  (void)g; (void)l;
#endif
}

__device__ __forceinline__ void wait_async0() {
#if defined(__AMDGCN__)
#if __has_builtin(__builtin_amdgcn_s_wait_asynccnt)
  __builtin_amdgcn_s_wait_asynccnt(0);
#else
  asm volatile("s_wait_asynccnt 0" ::: "memory");
#endif
#endif
}

__device__ __forceinline__ void wait_tensor0() {
#if defined(__AMDGCN__)
#if __has_builtin(__builtin_amdgcn_s_wait_tensorcnt)
  __builtin_amdgcn_s_wait_tensorcnt(0);
#else
  asm volatile("s_wait_tensorcnt 0" ::: "memory");
#endif
#endif
}

// LDS byte offset of a generic pointer into shared memory (AS3 pointers are
// raw LDS offsets on AMDGPU).
__device__ __forceinline__ unsigned lds_offset(const void* p) {
  return (unsigned)(size_t)(lv4i_p)(p);
}

// TDM: DMA a (rows=128) x (cols=32 bf16) tile, row stride = K elements, from
// global into LDS with 16B padding after every 64B row => 80B LDS row stride
// (= GLDT elements).  One wave issues for the whole workgroup; completion via
// TENSORcnt.
__device__ __forceinline__ void tdm_load_wtile(unsigned long long gaddr,
                                               unsigned lds_byte_off,
                                               int K, int N) {
#if USE_TDM
  v4u g0;
  g0.x = 1u;                                        // count=1 valid descriptor
  g0.y = lds_byte_off;                              // lds_addr
  g0.z = (unsigned)(gaddr & 0xffffffffu);           // global_addr[31:0]
  g0.w = (unsigned)((gaddr >> 32) & 0x01ffffffu) | (2u << 30);  // addr[56:32], type=2
  v8i g1;
  // word0: workgroup_mask=0 | data_size(2B)=1<<16 | pad_enable<<20 |
  //        pad_interval(16 DW)=3<<22 | pad_amount(4 DW)=3<<25
  g1[0] = (int)((1u << 16) | (1u << 20) | (3u << 22) | (3u << 25));
  g1[1] = (int)(((unsigned)K & 0xffffu) << 16);                 // tensor_dim0 lo
  g1[2] = (int)((((unsigned)K >> 16) & 0xffffu) |
                (((unsigned)N & 0xffffu) << 16));               // dim0 hi | dim1 lo
  g1[3] = (int)((((unsigned)N >> 16) & 0xffffu) | (32u << 16)); // dim1 hi | tile_dim0=32
  g1[4] = 128;                                                  // tile_dim1=128, tile_dim2=0
  g1[5] = K;                                                    // tensor_dim0_stride lo
  g1[6] = 0;                                                    // stride hi | dim1_stride lo
  g1[7] = 0;
  v4i z4 = {0, 0, 0, 0};                       // 2-D tensor: groups 2/3 unused
  v8i z8 = {0, 0, 0, 0, 0, 0, 0, 0};           // trailing group (6-arg form)
  __builtin_amdgcn_tensor_load_to_lds(g0, g1, z4, z4, z8, 0);
#else
  (void)gaddr; (void)lds_byte_off; (void)K; (void)N;
#endif
}

// ---------------------------------------------------------------- fp32 -> bf16
__global__ void cvt_bf16_kernel(const float* __restrict__ in,
                                unsigned short* __restrict__ out, size_t n) {
  size_t i = (size_t)blockIdx.x * blockDim.x + threadIdx.x;
  if (i < n) out[i] = f2bf(in[i]);
}

// ---------------------------------------------------------------- layernorm (D=1024), bf16 out
__global__ __launch_bounds__(256) void layernorm_kernel(
    const float* __restrict__ x, const float* __restrict__ w,
    const float* __restrict__ bias, unsigned short* __restrict__ out) {
  __shared__ float red[8];
  const int row = blockIdx.x, tid = threadIdx.x;
  const float* xr = x + (size_t)row * D_DIM;
  float lx[4];
#pragma unroll
  for (int i = 0; i < 4; ++i) lx[i] = xr[tid + i * 256];
  float s = lx[0] + lx[1] + lx[2] + lx[3];
#pragma unroll
  for (int d = 16; d; d >>= 1) s += __shfl_xor(s, d, 32);
  if ((tid & 31) == 0) red[tid >> 5] = s;
  __syncthreads();
  float tot = 0.f;
#pragma unroll
  for (int i = 0; i < 8; ++i) tot += red[i];
  const float mu = tot * (1.0f / D_DIM);
  __syncthreads();
  float vs = 0.f;
#pragma unroll
  for (int i = 0; i < 4; ++i) { float dq = lx[i] - mu; vs += dq * dq; }
#pragma unroll
  for (int d = 16; d; d >>= 1) vs += __shfl_xor(vs, d, 32);
  if ((tid & 31) == 0) red[tid >> 5] = vs;
  __syncthreads();
  float vtot = 0.f;
#pragma unroll
  for (int i = 0; i < 8; ++i) vtot += red[i];
  const float inv = rsqrtf(vtot * (1.0f / D_DIM) + EPS_F);
#pragma unroll
  for (int i = 0; i < 4; ++i) {
    int c = tid + i * 256;
    out[(size_t)row * D_DIM + c] = f2bf((lx[i] - mu) * inv * w[c] + bias[c]);
  }
}

// ---------------------------------------------------------------- GEMM: C = act(A @ W^T + bias)
// A: (M,K) bf16 row-major.  W: (N,K) bf16 row-major (so B = W^T is K x N).
// Block 128x128, BK=32, 256 threads = 8 waves in a 2x4 grid; wave owns 64x32.
#define GLDT 40  // LDS row stride (elements), 80B: 16B-aligned, bank-skewed

// One 32-K step of WMMAs out of an LDS tile pair.
__device__ __forceinline__ void gemm_tile_compute(
    const unsigned short* __restrict__ sAp, const unsigned short* __restrict__ sBp,
    v8f acc[4][2], int wm, int wn, int l16, int half) {
  v16bf afrag[4];
#pragma unroll
  for (int mt = 0; mt < 4; ++mt) {
    const unsigned short* pr = &sAp[(wm * 64 + mt * 16 + l16) * GLDT];
    afrag[mt] = load_frag16(pr + 16 * half, pr + 16 * half + 8);
  }
  v16bf bfrag[2];
#pragma unroll
  for (int nt = 0; nt < 2; ++nt) {
    const unsigned short* pr = &sBp[(wn * 32 + nt * 16 + l16) * GLDT + 16 * half];
    bfrag[nt] = load_frag16(pr, pr + 8);
  }
#pragma unroll
  for (int mt = 0; mt < 4; ++mt)
#pragma unroll
    for (int nt = 0; nt < 2; ++nt)
      acc[mt][nt] = wmma_bf16(afrag[mt], bfrag[nt], acc[mt][nt]);
}

template <bool RELU, bool OUT_BF16>
__global__ __launch_bounds__(256, 1) void gemm_bf16_kernel(
    const unsigned short* __restrict__ A, const unsigned short* __restrict__ W,
    const float* __restrict__ bias, void* __restrict__ Cout,
    int M, int N, int K) {
  const int tid = threadIdx.x;
  const int lane = tid & 31, wid = tid >> 5;
  const int l16 = lane & 15, half = lane >> 4;
  const int wm = wid >> 2, wn = wid & 3;  // 2 x 4 wave grid
  const int m0 = blockIdx.y * 128;
  const int n0 = blockIdx.x * 128;
  const int ldrow = tid >> 1;           // 0..127
  const int ldcol = (tid & 1) * 16;     // 0 or 16

  v8f acc[4][2];
#pragma unroll
  for (int mt = 0; mt < 4; ++mt)
#pragma unroll
    for (int nt = 0; nt < 2; ++nt)
#pragma unroll
      for (int i = 0; i < 8; ++i) acc[mt][nt][i] = 0.f;

  const unsigned short* gArow = A + (size_t)(m0 + ldrow) * K + ldcol;
  const unsigned short* gBrow = W + (size_t)(n0 + ldrow) * K + ldcol;

#if USE_ASYNC
  // Double-buffered LDS.  A tile: per-lane GLOBAL_LOAD_ASYNC_TO_LDS (ASYNCcnt).
  // B tile: one TENSOR_LOAD_TO_LDS descriptor issued by wave 0 (TENSORcnt),
  // padded by the TDM to the same bank-skewed layout.  DMA of tile t+1
  // overlaps the WMMAs on tile t.
  __shared__ __align__(16) unsigned short sA[2][128 * GLDT];
  __shared__ __align__(16) unsigned short sB[2][128 * GLDT];
  unsigned short* lA[2] = { &sA[0][ldrow * GLDT + ldcol], &sA[1][ldrow * GLDT + ldcol] };
#if !USE_TDM
  unsigned short* lB[2] = { &sB[0][ldrow * GLDT + ldcol], &sB[1][ldrow * GLDT + ldcol] };
#endif
  const unsigned long long wtile0 =
      (unsigned long long)(size_t)W + ((size_t)n0 * (size_t)K) * 2ull;
  const unsigned ldsB[2] = { lds_offset(&sB[0][0]), lds_offset(&sB[1][0]) };

  // prologue: tile 0 in flight
  async_b128(gArow + 0, lA[0]);
  async_b128(gArow + 8, lA[0] + 8);
#if USE_TDM
  if (wid == 0) tdm_load_wtile(wtile0, ldsB[0], K, N);
#else
  async_b128(gBrow + 0, lB[0]);
  async_b128(gBrow + 8, lB[0] + 8);
#endif

  int buf = 0;
  for (int k0 = 0; k0 < K; k0 += 32, buf ^= 1) {
    wait_async0();                 // my async A chunks landed in LDS
    wait_tensor0();                // wave-0's B tile DMA done (no-op elsewhere)
    __syncthreads();               // everyone's landed; prior compute finished
    if (k0 + 32 < K) {             // launch tile t+1 into the other buffer
      const int kn = k0 + 32;
      async_b128(gArow + kn,     lA[buf ^ 1]);
      async_b128(gArow + kn + 8, lA[buf ^ 1] + 8);
#if USE_TDM
      if (wid == 0) tdm_load_wtile(wtile0 + (size_t)kn * 2ull, ldsB[buf ^ 1], K, N);
#else
      async_b128(gBrow + kn,     lB[buf ^ 1]);
      async_b128(gBrow + kn + 8, lB[buf ^ 1] + 8);
#endif
    }
    gemm_tile_compute(sA[buf], sB[buf], acc, wm, wn, l16, half);
  }
#else
  // Fallback: register-staged software pipeline, single LDS buffer.
  __shared__ __align__(16) unsigned short sA1[128 * GLDT];
  __shared__ __align__(16) unsigned short sB1[128 * GLDT];
  uint4 ra0 = ((const uint4*)gArow)[0], ra1 = ((const uint4*)gArow)[1];
  uint4 rb0 = ((const uint4*)gBrow)[0], rb1 = ((const uint4*)gBrow)[1];
  for (int k0 = 0; k0 < K; k0 += 32) {
    __syncthreads();
    *(uint4*)&sA1[ldrow * GLDT + ldcol]     = ra0;
    *(uint4*)&sA1[ldrow * GLDT + ldcol + 8] = ra1;
    *(uint4*)&sB1[ldrow * GLDT + ldcol]     = rb0;
    *(uint4*)&sB1[ldrow * GLDT + ldcol + 8] = rb1;
    __syncthreads();
    if (k0 + 32 < K) {                 // next tile's loads overlap compute
      const int kn = k0 + 32;
      ra0 = ((const uint4*)(gArow + kn))[0];
      ra1 = ((const uint4*)(gArow + kn))[1];
      rb0 = ((const uint4*)(gBrow + kn))[0];
      rb1 = ((const uint4*)(gBrow + kn))[1];
    }
    gemm_tile_compute(sA1, sB1, acc, wm, wn, l16, half);
  }
#endif

#pragma unroll
  for (int nt = 0; nt < 2; ++nt) {
    const int col = n0 + wn * 32 + nt * 16 + l16;
    const float bv = bias ? bias[col] : 0.f;
#pragma unroll
    for (int mt = 0; mt < 4; ++mt) {
#pragma unroll
      for (int r = 0; r < 8; ++r) {
        const int row = m0 + wm * 64 + mt * 16 + r + 8 * half;
        float v = acc[mt][nt][r] + bv;
        if (RELU) v = fmaxf(v, 0.f);
        if (OUT_BF16)
          ((unsigned short*)Cout)[(size_t)row * N + col] = f2bf(v);
        else
          ((float*)Cout)[(size_t)row * N + col] = v;
      }
    }
  }
}

// ---------------------------------------------------------------- flash attention
// qkv: (T*B, 3*D) bf16, cols = g*1024 + h*64 + dk.  out: (T*B, D) bf16.
// Block = 128 threads (4 waves), grid = (T/128, B*H). Wave owns 32 q-rows.
#define ALDT 72  // LDS row stride (elems), 144B: 16B-aligned, bank-skewed

__global__ __launch_bounds__(128, 1) void attention_kernel(
    const unsigned short* __restrict__ qkv, unsigned short* __restrict__ out) {
  __shared__ __align__(16) unsigned short sK[64 * ALDT];       // (key, dk)
  __shared__ __align__(16) unsigned short sV[64 * ALDT];       // (dk, key)  transposed
  __shared__ __align__(16) unsigned short sP[4][32 * ALDT];    // per-wave P scratch

  const int tid = threadIdx.x, lane = tid & 31, w = tid >> 5;
  const int l16 = lane & 15, half = lane >> 4;
  const int qb = blockIdx.x;
  const int bh = blockIdx.y;
  const int b = bh / H_DIM, h = bh % H_DIM;
  const int qr0 = qb * 128 + w * 32;

  // Q fragments (registers for whole kernel)
  v16bf qf[2][2];
#pragma unroll
  for (int mt = 0; mt < 2; ++mt)
#pragma unroll
    for (int ks = 0; ks < 2; ++ks) {
      const int t = qr0 + mt * 16 + l16;
      const unsigned short* qp =
          qkv + (size_t)(t * BATCH + b) * (3 * D_DIM) + h * DK_DIM + ks * 32;
      qf[mt][ks] = load_frag16(qp + 16 * half, qp + 16 * half + 8);
    }

  float mi[2][8], li[2][8];
  v8f oacc[2][4];
#pragma unroll
  for (int mt = 0; mt < 2; ++mt) {
#pragma unroll
    for (int r = 0; r < 8; ++r) { mi[mt][r] = -INFINITY; li[mt][r] = 0.f; }
#pragma unroll
    for (int nt = 0; nt < 4; ++nt)
#pragma unroll
      for (int i = 0; i < 8; ++i) oacc[mt][nt][i] = 0.f;
  }

  const int kbE = (qb + 1) * 2;  // causal: key blocks of 64 up through q-block
  for (int kb = 0; kb < kbE; ++kb) {
    __syncthreads();
    // cooperative K / V^T staging: 512 chunks of 8 elems, 4 per thread.
    // K tile goes straight to LDS via async DMA; V needs a transpose so it
    // takes the VGPR path.
#pragma unroll
    for (int c = 0; c < 4; ++c) {
      const int chunk = tid + c * 128;
      const int row = chunk >> 3;             // key row 0..63
      const int off = (chunk & 7) * 8;        // dk 0..56
      const int t = kb * 64 + row;
      const unsigned short* kp =
          qkv + (size_t)(t * BATCH + b) * (3 * D_DIM) + D_DIM + h * DK_DIM + off;
#if USE_ASYNC
      async_b128(kp, &sK[row * ALDT + off]);
#else
      *(uint4*)&sK[row * ALDT + off] = *(const uint4*)kp;
#endif
      uint4 vv = *(const uint4*)(kp + D_DIM);  // V section
      const unsigned short* vsc = (const unsigned short*)&vv;
#pragma unroll
      for (int i = 0; i < 8; ++i) sV[(off + i) * ALDT + row] = vsc[i];
    }
#if USE_ASYNC
    wait_async0();
#endif
    __syncthreads();

#pragma unroll
    for (int mt = 0; mt < 2; ++mt) {
      v8f st[4];
#pragma unroll
      for (int nt = 0; nt < 4; ++nt) {
        v8f s;
#pragma unroll
        for (int i = 0; i < 8; ++i) s[i] = 0.f;
#pragma unroll
        for (int ks = 0; ks < 2; ++ks) {
          const unsigned short* bp =
              &sK[(nt * 16 + l16) * ALDT + ks * 32 + 16 * half];
          s = wmma_bf16(qf[mt][ks], load_frag16(bp, bp + 8), s);
        }
        const int key = kb * 64 + nt * 16 + l16;
#pragma unroll
        for (int r = 0; r < 8; ++r) {
          const int qrow = qr0 + mt * 16 + r + 8 * half;
          s[r] = (key > qrow) ? -INFINITY : s[r] * SCALE_F;
        }
        st[nt] = s;
      }
      // online-softmax row update (rows live across 16 lanes of each half)
#pragma unroll
      for (int r = 0; r < 8; ++r) {
        float mc = fmaxf(fmaxf(st[0][r], st[1][r]), fmaxf(st[2][r], st[3][r]));
#pragma unroll
        for (int d = 8; d; d >>= 1) mc = fmaxf(mc, __shfl_xor(mc, d, 32));
        const float mnew = fmaxf(mi[mt][r], mc);
        const float alpha = __expf(mi[mt][r] - mnew);
        float rs = 0.f;
#pragma unroll
        for (int nt = 0; nt < 4; ++nt) {
          float p = __expf(st[nt][r] - mnew);
          st[nt][r] = p;
          rs += p;
        }
#pragma unroll
        for (int d = 8; d; d >>= 1) rs += __shfl_xor(rs, d, 32);
        li[mt][r] = li[mt][r] * alpha + rs;
        mi[mt][r] = mnew;
#pragma unroll
        for (int nt = 0; nt < 4; ++nt) oacc[mt][nt][r] *= alpha;
      }
      // spill P (C-layout) -> LDS, to re-read in A-fragment layout
#pragma unroll
      for (int nt = 0; nt < 4; ++nt)
#pragma unroll
        for (int r = 0; r < 8; ++r)
          sP[w][(mt * 16 + r + 8 * half) * ALDT + nt * 16 + l16] =
              f2bf(st[nt][r]);
    }

    // O += P @ V   (wave-private LDS; DS ops are in-order within the wave)
#pragma unroll
    for (int mt = 0; mt < 2; ++mt)
#pragma unroll
      for (int ks = 0; ks < 2; ++ks) {
        const unsigned short* pp = &sP[w][(mt * 16 + l16) * ALDT + ks * 32];
        v16bf pf = load_frag16(pp + 16 * half, pp + 16 * half + 8);
#pragma unroll
        for (int nt = 0; nt < 4; ++nt) {
          const unsigned short* vp =
              &sV[(nt * 16 + l16) * ALDT + ks * 32 + 16 * half];
          oacc[mt][nt] = wmma_bf16(pf, load_frag16(vp, vp + 8), oacc[mt][nt]);
        }
      }
  }

  // normalize + store (t, b, h*64 + dk)
#pragma unroll
  for (int mt = 0; mt < 2; ++mt)
#pragma unroll
    for (int nt = 0; nt < 4; ++nt)
#pragma unroll
      for (int r = 0; r < 8; ++r) {
        const int t = qr0 + mt * 16 + r + 8 * half;
        const float v = oacc[mt][nt][r] / li[mt][r];
        out[(size_t)(t * BATCH + b) * D_DIM + h * DK_DIM + nt * 16 + l16] =
            f2bf(v);
      }
}

// ---------------------------------------------------------------- concat bf16: (rows, 2D) = [a | y]
__global__ void concat_kernel(const unsigned short* __restrict__ a,
                              const unsigned short* __restrict__ y,
                              unsigned short* __restrict__ out) {
  size_t idx = (size_t)blockIdx.x * blockDim.x + threadIdx.x;
  if (idx >= (size_t)NROWS * 2 * D_DIM) return;
  const size_t row = idx / (2 * D_DIM);
  const int c = (int)(idx % (2 * D_DIM));
  out[idx] = (c < D_DIM) ? a[row * D_DIM + c] : y[row * D_DIM + (c - D_DIM)];
}

// ---------------------------------------------------------------- GRU gate epilogue
__global__ void gru_kernel(const float* __restrict__ x,
                           const float* __restrict__ g,
                           const float* __restrict__ bg,
                           float* __restrict__ outf,
                           unsigned short* __restrict__ outbf) {
  size_t idx = (size_t)blockIdx.x * blockDim.x + threadIdx.x;
  if (idx >= (size_t)NROWS * D_DIM) return;
  const size_t row = idx / D_DIM;
  const int d = (int)(idx % D_DIM);
  const float* gr = g + row * (3 * D_DIM);
  const float r = 1.0f / (1.0f + __expf(-gr[d]));
  const float z = 1.0f / (1.0f + __expf(-(gr[D_DIM + d] - bg[d])));
  const float hh = tanhf(gr[2 * D_DIM + d] * r);
  const float o = (1.0f - z) * x[idx] + z * hh;
  outf[idx] = o;
  if (outbf) outbf[idx] = f2bf(o);
}

// ---------------------------------------------------------------- launch
extern "C" void kernel_launch(void* const* d_in, const int* in_sizes, int n_in,
                              void* d_out, int out_size, void* d_ws,
                              size_t ws_size, hipStream_t stream) {
  (void)in_sizes; (void)n_in; (void)out_size; (void)ws_size;
  const float* x      = (const float*)d_in[0];
  const float* w_qkv  = (const float*)d_in[1];
  const float* w_out  = (const float*)d_in[2];
  const float* b_out  = (const float*)d_in[3];
  const float* ln1_w  = (const float*)d_in[4];
  const float* ln1_b  = (const float*)d_in[5];
  const float* ln2_w  = (const float*)d_in[6];
  const float* ln2_b  = (const float*)d_in[7];
  const float* w_ff1  = (const float*)d_in[8];
  const float* b_ff1  = (const float*)d_in[9];
  const float* w_ff2  = (const float*)d_in[10];
  const float* b_ff2  = (const float*)d_in[11];
  const float* g1_w   = (const float*)d_in[12];
  const float* g1_bg  = (const float*)d_in[13];
  const float* g2_w   = (const float*)d_in[14];
  const float* g2_bg  = (const float*)d_in[15];
  float* outp = (float*)d_out;

  char* base = (char*)d_ws;
  size_t off = 0;
  auto alloc = [&](size_t bytes) -> void* {
    off = (off + 255) & ~(size_t)255;
    void* p = base + off;
    off += bytes;
    return p;
  };
  typedef unsigned short u16;
  u16* wqkv_bf = (u16*)alloc((size_t)3 * D_DIM * D_DIM * 2);
  u16* wout_bf = (u16*)alloc((size_t)D_DIM * D_DIM * 2);
  u16* g1w_bf  = (u16*)alloc((size_t)3 * D_DIM * 2 * D_DIM * 2);
  u16* g2w_bf  = (u16*)alloc((size_t)3 * D_DIM * 2 * D_DIM * 2);
  u16* wff1_bf = (u16*)alloc((size_t)DFF_DIM * D_DIM * 2);
  u16* wff2_bf = (u16*)alloc((size_t)D_DIM * DFF_DIM * 2);
  u16* x_bf    = (u16*)alloc((size_t)NROWS * D_DIM * 2);
  u16* ln_bf   = (u16*)alloc((size_t)NROWS * D_DIM * 2);
  u16* qkv_bf  = (u16*)alloc((size_t)NROWS * 3 * D_DIM * 2);
  u16* attn_bf = (u16*)alloc((size_t)NROWS * D_DIM * 2);
  u16* y_bf    = (u16*)alloc((size_t)NROWS * D_DIM * 2);
  u16* h1_bf   = (u16*)alloc((size_t)NROWS * D_DIM * 2);
  u16* cat_bf  = (u16*)alloc((size_t)NROWS * 2 * D_DIM * 2);
  float* g_f32 = (float*)alloc((size_t)NROWS * 3 * D_DIM * 4);
  float* h1_f32 = (float*)alloc((size_t)NROWS * D_DIM * 4);
  u16* ffa_bf  = (u16*)alloc((size_t)NROWS * DFF_DIM * 2);

  auto cvt = [&](const float* src, u16* dst, size_t n) {
    cvt_bf16_kernel<<<dim3((unsigned)((n + 255) / 256)), 256, 0, stream>>>(src, dst, n);
  };
  // weights -> bf16
  cvt(w_qkv, wqkv_bf, (size_t)3 * D_DIM * D_DIM);
  cvt(w_out, wout_bf, (size_t)D_DIM * D_DIM);
  cvt(g1_w,  g1w_bf,  (size_t)3 * D_DIM * 2 * D_DIM);
  cvt(g2_w,  g2w_bf,  (size_t)3 * D_DIM * 2 * D_DIM);
  cvt(w_ff1, wff1_bf, (size_t)DFF_DIM * D_DIM);
  cvt(w_ff2, wff2_bf, (size_t)D_DIM * DFF_DIM);
  cvt(x,     x_bf,    (size_t)NROWS * D_DIM);

  // ln1(x) -> bf16
  layernorm_kernel<<<NROWS, 256, 0, stream>>>(x, ln1_w, ln1_b, ln_bf);
  // qkv = ln1 @ w_qkv^T
  gemm_bf16_kernel<false, true><<<dim3(3 * D_DIM / 128, NROWS / 128), 256, 0, stream>>>(
      ln_bf, wqkv_bf, nullptr, qkv_bf, NROWS, 3 * D_DIM, D_DIM);
  // flash attention
  attention_kernel<<<dim3(T_DIM / 128, BATCH * H_DIM), 128, 0, stream>>>(qkv_bf, attn_bf);
  // out projection
  gemm_bf16_kernel<false, true><<<dim3(D_DIM / 128, NROWS / 128), 256, 0, stream>>>(
      attn_bf, wout_bf, b_out, y_bf, NROWS, D_DIM, D_DIM);
  // gru gate 1: g = cat(x, attn_proj) @ g1_w^T
  concat_kernel<<<dim3((unsigned)((size_t)NROWS * 2 * D_DIM / 256)), 256, 0, stream>>>(
      x_bf, y_bf, cat_bf);
  gemm_bf16_kernel<false, false><<<dim3(3 * D_DIM / 128, NROWS / 128), 256, 0, stream>>>(
      cat_bf, g1w_bf, nullptr, g_f32, NROWS, 3 * D_DIM, 2 * D_DIM);
  gru_kernel<<<dim3((unsigned)((size_t)NROWS * D_DIM / 256)), 256, 0, stream>>>(
      x, g_f32, g1_bg, h1_f32, h1_bf);
  // ln2(h1) -> bf16
  layernorm_kernel<<<NROWS, 256, 0, stream>>>(h1_f32, ln2_w, ln2_b, ln_bf);
  // ffn
  gemm_bf16_kernel<true, true><<<dim3(DFF_DIM / 128, NROWS / 128), 256, 0, stream>>>(
      ln_bf, wff1_bf, b_ff1, ffa_bf, NROWS, DFF_DIM, D_DIM);
  gemm_bf16_kernel<false, true><<<dim3(D_DIM / 128, NROWS / 128), 256, 0, stream>>>(
      ffa_bf, wff2_bf, b_ff2, y_bf, NROWS, D_DIM, DFF_DIM);
  // gru gate 2 -> output
  concat_kernel<<<dim3((unsigned)((size_t)NROWS * 2 * D_DIM / 256)), 256, 0, stream>>>(
      h1_bf, y_bf, cat_bf);
  gemm_bf16_kernel<false, false><<<dim3(3 * D_DIM / 128, NROWS / 128), 256, 0, stream>>>(
      cat_bf, g2w_bf, nullptr, g_f32, NROWS, 3 * D_DIM, 2 * D_DIM);
  gru_kernel<<<dim3((unsigned)((size_t)NROWS * D_DIM / 256)), 256, 0, stream>>>(
      h1_f32, g_f32, g2_bg, outp, nullptr);
}
